// PSM_4432406249586
// MI455X (gfx1250) — matrix operand order repaired and checked by reading, not compiled, not run on hardware
//
#include <hip/hip_runtime.h>

typedef __attribute__((ext_vector_type(16))) __bf16 v16bf;
typedef __attribute__((ext_vector_type(8)))  float  v8f;

#define HW_SZ 1024
#define C_SZ  16
#define K_SZ  4
#define WAVES 8
#define LOG2E 1.44269504088896340736f

static __device__ __forceinline__ float4 ldsF4(const float* p) {
    return *reinterpret_cast<const float4*>(p);
}

__global__ __launch_bounds__(256) void psm_rank1_attn_kernel(
    const float* __restrict__ feature,  // [B, C, HW]
    const float* __restrict__ clsmap,   // [B, K, HW]
    const float* __restrict__ gamma,    // [K]
    float* __restrict__ out)            // [B, C, HW]
{
    __shared__ float lds_f[HW_SZ];           // f[b,c,:]
    __shared__ float lds_v[K_SZ][HW_SZ];     // m[b,k,:] for all k
    __shared__ float lds_vmax[K_SZ];
    __shared__ float lds_vmin[K_SZ];

    const int tid  = threadIdx.x;
    const int lane = tid & 31;
    const int wave = tid >> 5;
    const int bc   = blockIdx.x;             // b*C + c
    const int b    = bc / C_SZ;
    const int ibase = blockIdx.y * (WAVES * 16) + wave * 16;

    // Stage f and all class maps into LDS.
    const float* fptr = feature + (size_t)bc * HW_SZ;
    for (int t = tid; t < HW_SZ; t += 256) lds_f[t] = fptr[t];
    const float* mptr = clsmap + (size_t)b * (K_SZ * HW_SZ);
    for (int t = tid; t < K_SZ * HW_SZ; t += 256) (&lds_v[0][0])[t] = mptr[t];
    __syncthreads();

    // Per-class min/max of v (rank-1 energy => analytic row max, single pass).
    if (wave < K_SZ) {
        float mx = -3.402823466e38f, mn = 3.402823466e38f;
        for (int t = lane; t < HW_SZ; t += 32) {
            float v = lds_v[wave][t];
            mx = fmaxf(mx, v);
            mn = fminf(mn, v);
        }
        #pragma unroll
        for (int off = 16; off > 0; off >>= 1) {
            mx = fmaxf(mx, __shfl_xor(mx, off, 32));
            mn = fminf(mn, __shfl_xor(mn, off, 32));
        }
        if (lane == 0) { lds_vmax[wave] = mx; lds_vmin[wave] = mn; }
    }
    __syncthreads();

    const bool hi    = lane >= 16;
    const int  col   = lane & 15;            // A-row index within tile; also B/D column
    const int  hoff8  = hi ? 8 : 0;
    const int  hoff16 = hi ? 16 : 0;
    const float s  = lds_f[ibase + col];     // query scalar for this row
    const float s2 = s * LOG2E;

    // Branchless B-column selectors: col0 -> f_j, col1 -> 1, else 0.
    const float c0 = (col == 0) ? 1.0f : 0.0f;
    const float c1 = (col == 1) ? 1.0f : 0.0f;

    // Per-class softmax shift (in log2 domain), hoisted out of the hot loop.
    float mx2[K_SZ];
    #pragma unroll
    for (int k = 0; k < K_SZ; ++k) {
        const float m = (s >= 0.0f) ? s * lds_vmax[k] : s * lds_vmin[k];
        mx2[k] = m * LOG2E;
    }

    v8f acc[K_SZ];
    #pragma unroll
    for (int k = 0; k < K_SZ; ++k) acc[k] = (v8f){};

    for (int jt = 0; jt < HW_SZ / 32; ++jt) {
        const int jbase = jt * 32;

        // ---- Stage ALL LDS data for this j-tile into distinct registers ----
        // (one load clause + one dscnt wait instead of load/wait/use chains)
        float4 fb[4];                         // B operand: f_j values
        #pragma unroll
        for (int q = 0; q < 4; ++q)
            fb[q] = ldsF4(&lds_f[jbase + hoff8 + 4 * (q & 1) + 16 * (q >> 1)]);

        float4 va[K_SZ][4];                   // A operand: v_j values per class
        #pragma unroll
        for (int k = 0; k < K_SZ; ++k) {
            const float* vp = &lds_v[k][jbase + hoff8];
            va[k][0] = ldsF4(vp + 0);
            va[k][1] = ldsF4(vp + 4);
            va[k][2] = ldsF4(vp + 16);
            va[k][3] = ldsF4(vp + 20);
        }

        // ---- B: 32x16 bf16, col0 = f_j, col1 = 1, rest 0 (branchless) ----
        v16bf bm;
        #pragma unroll
        for (int q = 0; q < 4; ++q) {
            bm[4 * q + 0] = (__bf16)(fb[q].x * c0 + c1);
            bm[4 * q + 1] = (__bf16)(fb[q].y * c0 + c1);
            bm[4 * q + 2] = (__bf16)(fb[q].z * c0 + c1);
            bm[4 * q + 3] = (__bf16)(fb[q].w * c0 + c1);
        }

        // ---- A tiles for all 4 classes: exp2(s2*v - mx2), dense VALU block ----
        v16bf a[K_SZ];
        #pragma unroll
        for (int k = 0; k < K_SZ; ++k) {
            const float m2 = mx2[k];
            #pragma unroll
            for (int q = 0; q < 4; ++q) {
                a[k][4 * q + 0] = (__bf16)__builtin_amdgcn_exp2f(s2 * va[k][q].x - m2);
                a[k][4 * q + 1] = (__bf16)__builtin_amdgcn_exp2f(s2 * va[k][q].y - m2);
                a[k][4 * q + 2] = (__bf16)__builtin_amdgcn_exp2f(s2 * va[k][q].z - m2);
                a[k][4 * q + 3] = (__bf16)__builtin_amdgcn_exp2f(s2 * va[k][q].w - m2);
            }
        }

        // ---- 4 independent WMMAs back-to-back (hazards covered by issue gap) ----
        #pragma unroll
        for (int k = 0; k < K_SZ; ++k) {
            acc[k] = __builtin_amdgcn_wmma_f32_16x16x32_bf16(
                false, a[k], false, bm, (short)0, acc[k], false, false);
        }
    }

    // D layout: lane n<16 -> column n, VGPR r -> row r; lane n>=16 -> row 8+r.
    // column 0 = numerator, column 1 = denominator.
    float res[8];
    #pragma unroll
    for (int r = 0; r < 8; ++r) res[r] = 0.0f;

    #pragma unroll
    for (int k = 0; k < K_SZ; ++k) {
        const float gk = gamma[k];
        #pragma unroll
        for (int r = 0; r < 8; ++r) {
            const float num = acc[k][r];
            const float den = __shfl_xor(acc[k][r], 1, 32);
            res[r] += gk * (num / den);
        }
    }

    // Lanes 0 and 16 hold valid results for rows 0-7 and 8-15 of the tile.
    if (col == 0) {
        const int rb = ibase + (hi ? 8 : 0);
        #pragma unroll
        for (int r = 0; r < 8; ++r) {
            const int i = rb + r;
            out[(size_t)bc * HW_SZ + i] = lds_f[i] + res[r];
        }
    }
}

extern "C" void kernel_launch(void* const* d_in, const int* in_sizes, int n_in,
                              void* d_out, int out_size, void* d_ws, size_t ws_size,
                              hipStream_t stream) {
    const float* feature = (const float*)d_in[0];
    const float* clsmap  = (const float*)d_in[1];
    const float* gamma   = (const float*)d_in[2];
    float* out = (float*)d_out;

    const int B = in_sizes[0] / (C_SZ * HW_SZ);   // = 2 for the reference shapes

    dim3 grid(B * C_SZ, HW_SZ / (16 * WAVES));    // (32, 8)
    psm_rank1_attn_kernel<<<grid, 256, 0, stream>>>(feature, clsmap, gamma, out);
}
